// EnhancedLSTM_86741159510271
// MI455X (gfx1250) — matrix operand order repaired and checked by reading, not compiled, not run on hardware
//
#include <hip/hip_runtime.h>
#include <hip/hip_bf16.h>

#define VOCAB 32000
#define EMB   512
#define HID   512
#define GATES 2048          // 4*HID
#define BSZ   16
#define SEQ   256
#define MALL  4096          // BSZ*SEQ
#define KDIM  512
#define KT    16            // KDIM/32
#define NBLK_LSTM 16

typedef __attribute__((ext_vector_type(16))) _Float16 v16h;
typedef __attribute__((ext_vector_type(8)))  _Float16 v8h;
typedef __attribute__((ext_vector_type(8)))  float    v8f;
typedef __attribute__((ext_vector_type(4)))  unsigned int uint32x4;
typedef __attribute__((ext_vector_type(8)))  int int32x8;
typedef __attribute__((ext_vector_type(4)))  int int32x4;

// ---------------------------------------------------------------- TDM copy
// 1D DMA of `bytes` (multiple of 8, n8 <= 65535 elems) global -> LDS.
// D# per CDNA5 ISA ch.8: group0 {count=1, lds_addr, global_addr, type=2},
// group1 {data_size=8B, tensor_dim0=n8, tensor_dim1=1, tile_dim0=n8,
// tile_dim1=1, tensor_dim0_stride=n8}.  Issue from ONE wave per block.
__device__ __forceinline__ void tdm_copy_1d(void* lds_ptr, const void* gptr,
                                            unsigned bytes) {
  unsigned n8 = bytes >> 3;
  unsigned lds_off = (unsigned)(size_t)lds_ptr;          // addr[31:0] = LDS offset
  unsigned long long ga = (unsigned long long)(size_t)gptr;
  uint32x4 g0 = { 1u,                                    // count = 1
                  lds_off,
                  (unsigned)ga,
                  ((unsigned)(ga >> 32) & 0x01FFFFFFu) | (2u << 30) }; // type=2
  int32x8 g1 = { (int)(3u << 16),                        // data_size = 8B
                 (int)((n8 & 0xFFFFu) << 16),            // tensor_dim0 lo
                 (int)(((n8 >> 16) & 0xFFFFu) | (1u << 16)), // dim0 hi, dim1=1
                 (int)((n8 & 0xFFFFu) << 16),            // tile_dim0 = n8
                 1,                                      // tile_dim1 = 1
                 (int)n8,                                // tensor_dim0_stride
                 0, 0 };
  int32x4 z4 = { 0, 0, 0, 0 };
  int32x8 z8 = { 0, 0, 0, 0, 0, 0, 0, 0 };
  __builtin_amdgcn_tensor_load_to_lds(g0, g1, z4, z4, z8, 0);
}

// ---------------------------------------------------------------- fragments
__device__ __forceinline__ v16h load_a_frag(const _Float16* base, int ld,
                                            int k0, int lane) {
  const int m = lane & 15;
  const int k = k0 + ((lane >> 4) << 3);
  const _Float16* p = base + (size_t)m * ld + k;
  v8h lo = *(const v8h*)p;
  v8h hi = *(const v8h*)(p + 16);
  v16h r;
#pragma unroll
  for (int i = 0; i < 8; ++i) { r[i] = lo[i]; r[i + 8] = hi[i]; }
  return r;
}

__device__ __forceinline__ v16h load_b_frag(const _Float16* F, size_t fragIdx,
                                            int lane) {
  return ((const v16h*)(F + fragIdx * 512))[lane];
}

__device__ __forceinline__ float sigf(float x) { return 1.f / (1.f + __expf(-x)); }

// ---------------------------------------------------------------- init
__global__ void k_init(unsigned* bar) {
  if (threadIdx.x < 2) bar[threadIdx.x] = 0u;
}

// ---------------------------------------------------------------- embedding
__global__ void k_embed(const int* __restrict__ x, const float* __restrict__ emb,
                        _Float16* __restrict__ XE) {
  int idx = blockIdx.x * 256 + threadIdx.x;
  int m = idx >> 9, e = idx & 511;
  int b = m & 15, t = m >> 4;
  int tok = x[b * SEQ + t];
  XE[(size_t)m * EMB + e] = (_Float16)emb[(size_t)tok * EMB + e];
}

// ---------------------------------------------------------------- weight swizzle
__global__ void k_pack(const float* __restrict__ W, _Float16* __restrict__ F,
                       int N) {
  int idx = blockIdx.x * 256 + threadIdx.x;
  int n = idx >> 9, k = idx & 511;
  if (n >= N) return;
  int nt = n >> 4, ln = n & 15;
  int kt = k >> 5, kk = k & 31;
  int hi = (kk >> 3) & 1, e = (kk & 7) + ((kk >> 4) << 3);
  int lane = ln + (hi << 4);
  size_t fragIdx = (size_t)nt * KT + kt;
  F[(fragIdx * 32 + lane) * 16 + e] = (_Float16)W[(size_t)n * KDIM + k];
}

// ---------------------------------------------------------------- WMMA GEMM
// Block = 128(M) x 64(N) tile, 8 waves stacked in M sharing one B panel.
// B panel (4 N-tiles x 16 K-steps = 64 KB, contiguous fragments) is staged
// into LDS with one TDM DMA, then each wave streams A from global.
__global__ void k_gemm(const _Float16* __restrict__ A,
                       const _Float16* __restrict__ BF,
                       const float* __restrict__ bias,
                       float* __restrict__ out, int M, int N, int mode) {
  __shared__ __align__(16) _Float16 sB[4 * KT * 512];   // 64 KB
  const int lane = threadIdx.x & 31;
  const int wave = threadIdx.x >> 5;
  const int nq64 = N >> 6;
  const int mb = blockIdx.x / nq64;
  const int nq = blockIdx.x % nq64;

  if (threadIdx.x < 32) {
    tdm_copy_1d(sB, BF + (size_t)(nq * 4) * KT * 512, 4 * KT * 512 * 2);
    __builtin_amdgcn_s_wait_tensorcnt(0);
  }
  __syncthreads();

  const int mt = mb * 8 + wave;
  v8f c0 = {}, c1 = {}, c2 = {}, c3 = {};
  const _Float16* Arow = A + (size_t)mt * 16 * KDIM;
#pragma unroll 4
  for (int kt = 0; kt < KT; ++kt) {
    v16h a = load_a_frag(Arow, KDIM, kt * 32, lane);
    if (kt + 1 < KT)
      __builtin_prefetch((const void*)(Arow + (kt + 1) * 32), 0, 1);
    v16h b0 = load_b_frag(sB, (size_t)kt, lane);
    v16h b1 = load_b_frag(sB, (size_t)KT + kt, lane);
    v16h b2 = load_b_frag(sB, (size_t)2 * KT + kt, lane);
    v16h b3 = load_b_frag(sB, (size_t)3 * KT + kt, lane);
    c0 = __builtin_amdgcn_wmma_f32_16x16x32_f16(false, a, false, b0, (short)0, c0, false, false);
    c1 = __builtin_amdgcn_wmma_f32_16x16x32_f16(false, a, false, b1, (short)0, c1, false, false);
    c2 = __builtin_amdgcn_wmma_f32_16x16x32_f16(false, a, false, b2, (short)0, c2, false, false);
    c3 = __builtin_amdgcn_wmma_f32_16x16x32_f16(false, a, false, b3, (short)0, c3, false, false);
  }

  const int ln = lane & 15, hi = lane >> 4;
  v8f acc[4] = {c0, c1, c2, c3};
#pragma unroll
  for (int j = 0; j < 4; ++j) {
    int n = (nq * 4 + j) * 16 + ln;
    float bv = bias[n];
#pragma unroll
    for (int r = 0; r < 8; ++r) {
      int m = mt * 16 + r + hi * 8;
      float v = acc[j][r] + bv;
      if (mode == 0) {
        out[(size_t)m * N + n] = v;
      } else {
        int b = m & 15, t = m >> 4;
        out[((size_t)b * SEQ + t) * (size_t)N + n] = v;
      }
    }
  }
}

// ---------------------------------------------------------------- grid barrier
__device__ __forceinline__ void grid_barrier(unsigned* counter,
                                             volatile unsigned* gen) {
  __syncthreads();
  if (threadIdx.x == 0) {
    __threadfence();
    unsigned my = *gen;
    unsigned old = atomicAdd(counter, 1u);
    if (old == NBLK_LSTM - 1u) {
      *counter = 0u;
      __threadfence();
      atomicAdd((unsigned*)gen, 1u);
    } else {
      while (*gen == my) { __builtin_amdgcn_s_sleep(1); }
    }
    __threadfence();
  }
  __syncthreads();
}

// ---------------------------------------------------------------- LSTM layer
__global__ void k_lstm(const float* __restrict__ G,
                       const _Float16* __restrict__ WhhF,
                       _Float16* __restrict__ Hout,
                       _Float16* __restrict__ hbuf,
                       unsigned* __restrict__ bar) {
  __shared__ __align__(16) _Float16 sH[16 * KDIM];   // 16 KB
  __shared__ __align__(16) float    sGate[16 * 128]; //  8 KB
  __shared__ __align__(16) float    sC[16 * 32];     //  2 KB

  const int wg   = blockIdx.x;
  const int wave = threadIdx.x >> 5;
  const int lane = threadIdx.x & 31;
  const int ln = lane & 15, hi = lane >> 4;

  const int g   = wave >> 1;
  const int sub = wave & 1;
  const int gnt = g * 32 + wg * 2 + sub;   // N-tile index in 2048

  for (int t = 0; t < SEQ; ++t) {
    // ---- stage h into LDS (TDM broadcast read of the shared h vector)
    if (t == 0) {
      for (int i = threadIdx.x; i < 16 * KDIM; i += 256) sH[i] = (_Float16)0.f;
      for (int i = threadIdx.x; i < 16 * 32; i += 256) sC[i] = 0.f;
    } else if (threadIdx.x < 32) {
      tdm_copy_1d(sH, hbuf, 16 * KDIM * 2);
      __builtin_amdgcn_s_wait_tensorcnt(0);
    }
    __syncthreads();

    // ---- accumulator = precomputed input projection G (bias folded in)
    v8f acc;
    {
      const float* gp = G + ((size_t)(t * 16 + hi * 8)) * GATES + gnt * 16 + ln;
#pragma unroll
      for (int r = 0; r < 8; ++r) acc[r] = gp[(size_t)r * GATES];
    }

    // ---- h @ Whh^T : 16 WMMA K-steps (B frags stream from L2)
#pragma unroll 4
    for (int kt = 0; kt < KT; ++kt) {
      v16h a = load_a_frag(sH, KDIM, kt * 32, lane);
      size_t fi = (size_t)gnt * KT + kt;
      if (kt + 1 < KT)
        __builtin_prefetch((const void*)(WhhF + (fi + 1) * 512), 0, 1);
      v16h b = load_b_frag(WhhF, fi, lane);
      acc = __builtin_amdgcn_wmma_f32_16x16x32_f16(false, a, false, b, (short)0, acc, false, false);
    }

#pragma unroll
    for (int r = 0; r < 8; ++r)
      sGate[(r + hi * 8) * 128 + wave * 16 + ln] = acc[r];
    __syncthreads();

    // ---- pointwise LSTM cell update for this block's 32 columns
    for (int e = threadIdx.x; e < 16 * 32; e += 256) {
      int rb = e >> 5, jc = e & 31;
      float iv = sGate[rb * 128 + jc];
      float fv = sGate[rb * 128 + 32 + jc];
      float gv = sGate[rb * 128 + 64 + jc];
      float ov = sGate[rb * 128 + 96 + jc];
      float cp = (t == 0) ? 0.f : sC[e];
      float cn = sigf(fv) * cp + sigf(iv) * tanhf(gv);
      float hn = sigf(ov) * tanhf(cn);
      sC[e] = cn;
      _Float16 hh = (_Float16)hn;
      hbuf[(size_t)rb * KDIM + wg * 32 + jc] = hh;
      Hout[((size_t)t * 16 + rb) * KDIM + wg * 32 + jc] = hh;
    }
    __threadfence();
    grid_barrier(bar, bar + 1);
  }
}

// ---------------------------------------------------------------- launch
extern "C" void kernel_launch(void* const* d_in, const int* in_sizes, int n_in,
                              void* d_out, int out_size, void* d_ws, size_t ws_size,
                              hipStream_t stream) {
  const int*   x    = (const int*)d_in[0];
  const float* emb  = (const float*)d_in[1];
  const float* Wih  = (const float*)d_in[2];   // (2, 2048, 512)
  const float* Whh  = (const float*)d_in[3];   // (2, 2048, 512)
  const float* bias = (const float*)d_in[4];   // (2, 2048)
  const float* fcw  = (const float*)d_in[5];   // (32000, 512)
  const float* fcb  = (const float*)d_in[6];   // (32000,)
  float* logits = (float*)d_out;

  char* ws = (char*)d_ws;
  size_t off = 0;
  auto alloc = [&](size_t bytes) {
    size_t o = off;
    off = (off + bytes + 255) & ~(size_t)255;
    return o;
  };
  _Float16* XE    = (_Float16*)(ws + alloc((size_t)MALL * EMB * 2));
  _Float16* H0    = (_Float16*)(ws + alloc((size_t)MALL * HID * 2));
  _Float16* H1    = (_Float16*)(ws + alloc((size_t)MALL * HID * 2));
  _Float16* WihF0 = (_Float16*)(ws + alloc((size_t)GATES * KDIM * 2));
  _Float16* WihF1 = (_Float16*)(ws + alloc((size_t)GATES * KDIM * 2));
  _Float16* WhhF0 = (_Float16*)(ws + alloc((size_t)GATES * KDIM * 2));
  _Float16* WhhF1 = (_Float16*)(ws + alloc((size_t)GATES * KDIM * 2));
  _Float16* FCF   = (_Float16*)(ws + alloc((size_t)VOCAB * KDIM * 2));
  float*    G     = (float*)(ws + alloc((size_t)MALL * GATES * 4));
  _Float16* hbuf  = (_Float16*)(ws + alloc((size_t)BSZ * HID * 2));
  unsigned* bar   = (unsigned*)(ws + alloc(256));
  (void)ws_size; (void)in_sizes; (void)n_in; (void)out_size;

  k_init<<<1, 64, 0, stream>>>(bar);
  k_embed<<<(MALL * EMB) / 256, 256, 0, stream>>>(x, emb, XE);

  k_pack<<<(GATES * KDIM) / 256, 256, 0, stream>>>(Wih, WihF0, GATES);
  k_pack<<<(GATES * KDIM) / 256, 256, 0, stream>>>(Wih + (size_t)GATES * KDIM, WihF1, GATES);
  k_pack<<<(GATES * KDIM) / 256, 256, 0, stream>>>(Whh, WhhF0, GATES);
  k_pack<<<(GATES * KDIM) / 256, 256, 0, stream>>>(Whh + (size_t)GATES * KDIM, WhhF1, GATES);
  k_pack<<<(VOCAB * KDIM) / 256, 256, 0, stream>>>(fcw, FCF, VOCAB);

  // layer 0: batched input projection (bias folded), then recurrence
  k_gemm<<<(MALL / 128) * (GATES / 64), 256, 0, stream>>>(XE, WihF0, bias, G, MALL, GATES, 0);
  k_lstm<<<NBLK_LSTM, 256, 0, stream>>>(G, WhhF0, H0, hbuf, bar);

  // layer 1
  k_gemm<<<(MALL / 128) * (GATES / 64), 256, 0, stream>>>(H0, WihF1, bias + GATES, G, MALL, GATES, 0);
  k_lstm<<<NBLK_LSTM, 256, 0, stream>>>(G, WhhF1, H1, hbuf, bar);

  // final vocab projection, permuted store to (B, S, V)
  k_gemm<<<(MALL / 128) * (VOCAB / 64), 256, 0, stream>>>(H1, FCF, fcb, logits, MALL, VOCAB, 1);
}
